// SelfAttentionDiff_23802708754714
// MI455X (gfx1250) — matrix operand order repaired and checked by reading, not compiled, not run on hardware
//
#include <hip/hip_runtime.h>
#include <stdint.h>

#define BATCH 8
#define CH    128
#define NPIX  4096
#define NGRP  32
#define GEPS  1e-5f

typedef float        v8f   __attribute__((ext_vector_type(8)));
typedef unsigned int v8u   __attribute__((ext_vector_type(8)));
typedef __bf16       v16bf __attribute__((ext_vector_type(16)));
typedef unsigned short bf16_t;

// ---------------- helpers ----------------

__device__ __forceinline__ bf16_t f2bf(float f) {
    unsigned int u = __float_as_uint(f);
    unsigned int r = u + 0x7fffu + ((u >> 16) & 1u);   // round-to-nearest-even
    return (bf16_t)(r >> 16);
}

// A fragment: 16x32 bf16, row-major source, element (r,k) at p[r*ld + k].
// Layout: lane r=lane&15, h=lane>>4; VGPR v: k=2v+16h (matches B fragments)
__device__ __forceinline__ v16bf load_a(const bf16_t* p, int ld, int lane) {
    int r = lane & 15, h = lane >> 4;
    const bf16_t* row = p + r * ld + 16 * h;
    v8u u;
#pragma unroll
    for (int v = 0; v < 4; ++v) u[v]     = *(const unsigned int*)(row + 2 * v);
#pragma unroll
    for (int v = 0; v < 4; ++v) u[4 + v] = *(const unsigned int*)(row + 8 + 2 * v);
    return __builtin_bit_cast(v16bf, u);
}

// B fragment: 32x16 bf16 from an N-major ("NxK") source: element (k,n) at p[n*ld + k].
// Layout: n=lane&15, h=lane>>4; VGPR v: k=2v+16h (pair contiguous -> dword load)
__device__ __forceinline__ v16bf load_b_nk(const bf16_t* p, int ld, int lane) {
    int n = lane & 15, h = lane >> 4;
    const bf16_t* col = p + n * ld + 16 * h;
    v8u u;
#pragma unroll
    for (int v = 0; v < 8; ++v) u[v] = *(const unsigned int*)(col + 2 * v);
    return __builtin_bit_cast(v16bf, u);
}

// B fragment from a K-major ("KxN") source: element (k,n) at p[k*ld + n].
__device__ __forceinline__ v16bf load_b_kn(const bf16_t* p, int ld, int lane) {
    int n = lane & 15, h = lane >> 4;
    v8u u;
#pragma unroll
    for (int v = 0; v < 8; ++v) {
        unsigned int lo = p[(2 * v + 16 * h) * ld + n];
        unsigned int hi = p[(2 * v + 16 * h + 1) * ld + n];
        u[v] = lo | (hi << 16);
    }
    return __builtin_bit_cast(v16bf, u);
}

__device__ __forceinline__ v8f wmma_bf16(v16bf a, v16bf b, v8f c) {
    return __builtin_amdgcn_wmma_f32_16x16x32_bf16(false, a, false, b, (short)0, c, false, false);
}

__device__ __forceinline__ float hmax16(float v) {
#pragma unroll
    for (int m = 1; m <= 8; m <<= 1) v = fmaxf(v, __shfl_xor(v, m, 16));
    return v;
}
__device__ __forceinline__ float hsum16(float v) {
#pragma unroll
    for (int m = 1; m <= 8; m <<= 1) v += __shfl_xor(v, m, 16);
    return v;
}

// Generic (flat) LDS addresses carry the LDS byte offset in the low 32 bits
// (aperture bits live in [63:32]); truncation yields the VDST operand for
// GLOBAL_LOAD_ASYNC_TO_LDS.
__device__ __forceinline__ unsigned lds_off(const void* p) {
    return (unsigned)(uintptr_t)p;
}

// CDNA5 async global->LDS copy, 16 bytes per lane; tracked by ASYNCcnt.
__device__ __forceinline__ void async_copy_b128(unsigned lds_byte_off, const void* gaddr) {
    asm volatile("global_load_async_to_lds_b128 %0, %1, off"
                 :: "v"(lds_byte_off), "v"(gaddr)
                 : "memory");
}
__device__ __forceinline__ void wait_asynccnt0() {
    asm volatile("s_wait_asynccnt 0x0" ::: "memory");
}

// ---------------- kernel 1: GroupNorm statistics ----------------
// one block per (b, g); the 4 channels of a group are contiguous (4*4096 floats)
__global__ void gn_stats_kernel(const float* __restrict__ x, float* __restrict__ stats) {
    int bg = blockIdx.x;
    const float* base = x + (size_t)bg * 4 * NPIX;
    float s = 0.f, sq = 0.f;
    for (int i = threadIdx.x; i < 4 * NPIX; i += 256) {
        float v = base[i];
        s += v; sq += v * v;
    }
    __shared__ float rs[256], rq[256];
    rs[threadIdx.x] = s; rq[threadIdx.x] = sq;
    __syncthreads();
    for (int off = 128; off > 0; off >>= 1) {
        if (threadIdx.x < off) { rs[threadIdx.x] += rs[threadIdx.x + off]; rq[threadIdx.x] += rq[threadIdx.x + off]; }
        __syncthreads();
    }
    if (threadIdx.x == 0) {
        float mean = rs[0] * (1.f / 16384.f);
        float var  = rq[0] * (1.f / 16384.f) - mean * mean;
        stats[bg * 2 + 0] = mean;
        stats[bg * 2 + 1] = rsqrtf(var + GEPS);
    }
}

// ---------------- kernel 2: GN apply + transpose to (b, n, c) bf16 ----------------
__global__ void gn_apply_kernel(const float* __restrict__ x, const float* __restrict__ gw,
                                const float* __restrict__ gb, const float* __restrict__ stats,
                                bf16_t* __restrict__ xnT) {
    __shared__ bf16_t tile[32 * 128];
    int i0 = blockIdx.x * 32;
    int b  = blockIdx.y;
    const float* xb = x + (size_t)b * CH * NPIX;
    // read coalesced along i, normalize, stash in LDS as [i][c]
    for (int it = 0; it < 16; ++it) {
        int idx = it * 256 + threadIdx.x;
        int ii = idx & 31, c = idx >> 5;
        int g = c >> 2;
        float mean = stats[(b * NGRP + g) * 2 + 0];
        float rstd = stats[(b * NGRP + g) * 2 + 1];
        float v = xb[(size_t)c * NPIX + i0 + ii];
        v = (v - mean) * rstd * gw[c] + gb[c];
        tile[ii * 128 + c] = f2bf(v);
    }
    __syncthreads();
    // write coalesced along c
    bf16_t* ob = xnT + (size_t)b * NPIX * CH;
    for (int it = 0; it < 16; ++it) {
        int idx = it * 256 + threadIdx.x;
        int c = idx & 127, ii = idx >> 7;
        ob[(size_t)(i0 + ii) * CH + c] = tile[ii * 128 + c];
    }
}

// ---------------- kernel 3: weight f32 -> bf16 ----------------
__global__ void w2bf_kernel(const float* __restrict__ src, bf16_t* __restrict__ dst, int n) {
    int i = blockIdx.x * 256 + threadIdx.x;
    if (i < n) dst[i] = f2bf(src[i]);
}

// ---------------- kernel 4: projection GEMM ----------------
// out[i][co] = sum_k xnT[i][k] * W[co][k] + bias[co]   (out bf16, layout (b,n,c))
__global__ void proj_kernel(const bf16_t* __restrict__ xnT, const bf16_t* __restrict__ Wb,
                            const float* __restrict__ bias, bf16_t* __restrict__ outT) {
    int lane = threadIdx.x & 31;
    int wave = threadIdx.x >> 5;
    int i0  = blockIdx.x * 64 + wave * 16;
    int co0 = blockIdx.y * 64;

    v16bf a[4];
#pragma unroll
    for (int kk = 0; kk < 4; ++kk)
        a[kk] = load_a(xnT + (size_t)i0 * CH + kk * 32, CH, lane);

    int n = lane & 15, h = lane >> 4;
#pragma unroll
    for (int nt = 0; nt < 4; ++nt) {
        v8f acc = {};
#pragma unroll
        for (int kk = 0; kk < 4; ++kk) {
            v16bf bfr = load_b_nk(Wb + (size_t)(co0 + nt * 16) * CH + kk * 32, CH, lane);
            acc = wmma_bf16(a[kk], bfr, acc);
        }
        int col = co0 + nt * 16 + n;
        float bc = bias[col];
#pragma unroll
        for (int v = 0; v < 8; ++v) {
            int row = i0 + v + 8 * h;
            outT[(size_t)row * CH + col] = f2bf(acc[v] + bc);
        }
    }
}

// ---------------- kernel 5: fused flash attention ----------------
// qT,kT,vT: (b, n, c) bf16.   oT: (b, n, c) bf16.
// block = 128 threads (4 waves); BM=64 queries per block, BN=64 keys per tile.
// K/V tiles are staged into LDS with CDNA5 async copies (GLOBAL_LOAD_ASYNC_TO_LDS_B128),
// next tile is warmed with global_prefetch while WMMAs run.
__global__ void attn_kernel(const bf16_t* __restrict__ qT, const bf16_t* __restrict__ kT,
                            const bf16_t* __restrict__ vT, bf16_t* __restrict__ oT) {
    __shared__ __align__(16) bf16_t kt[64 * 128];   // [j][c]
    __shared__ __align__(16) bf16_t vt[64 * 128];   // [j][c]
    __shared__ __align__(16) bf16_t pl[64 * 64];    // [i][j] per-block P staging

    int lane = threadIdx.x & 31;
    int wave = threadIdx.x >> 5;
    int b = blockIdx.y;
    size_t base = (size_t)b * NPIX * CH;
    int i0 = blockIdx.x * 64;
    int iw = i0 + wave * 16;

    // per-thread tile-copy coordinates: 1024 16B-chunks per tile, 8 per thread
    int trow0 = threadIdx.x >> 4;          // 0..7   (row stride 8)
    int tc4   = threadIdx.x & 15;          // 16B chunk within a 256B row

    // preload this wave's Q fragments (16 rows x 128 k)
    v16bf aq[4];
#pragma unroll
    for (int kk = 0; kk < 4; ++kk)
        aq[kk] = load_a(qT + base + (size_t)iw * CH + kk * 32, CH, lane);

    v8f o[8];
#pragma unroll
    for (int t = 0; t < 8; ++t) o[t] = (v8f){};
    float m8[8], l8[8];
#pragma unroll
    for (int v = 0; v < 8; ++v) { m8[v] = -1e30f; l8[v] = 0.f; }

    const float sc = 0.08838834764831845f * 1.4426950408889634f; // (1/sqrt(128))*log2(e)
    int n = lane & 15, h = lane >> 4;

    for (int j0 = 0; j0 < NPIX; j0 += 64) {
        // async global->LDS staging of K/V tiles (64 rows x 128 bf16 each)
#pragma unroll
        for (int it = 0; it < 8; ++it) {
            int row = trow0 + it * 8;
            unsigned doff = (unsigned)(row * 128 + tc4 * 8) * sizeof(bf16_t);
            size_t goff = base + (size_t)(j0 + row) * CH + tc4 * 8;
            async_copy_b128(lds_off(kt) + doff, (const void*)(kT + goff));
            async_copy_b128(lds_off(vt) + doff, (const void*)(vT + goff));
        }
        // warm the next tile into L2/WGP$ while this one is consumed
        if (j0 + 64 < NPIX) {
            int prow = threadIdx.x >> 1;                      // 0..63
            int phalf = (threadIdx.x & 1) * 64;               // two 128B halves per row
            size_t poff = base + (size_t)(j0 + 64 + prow) * CH + phalf;
            __builtin_prefetch(kT + poff, 0, 1);
            __builtin_prefetch(vT + poff, 0, 1);
        }
        wait_asynccnt0();
        __syncthreads();

        // S = Q K^T  (4 column tiles of 16 keys)
        v8f s[4];
#pragma unroll
        for (int nt = 0; nt < 4; ++nt) {
            v8f acc = {};
#pragma unroll
            for (int kk = 0; kk < 4; ++kk) {
                v16bf bfr = load_b_nk(kt + (nt * 16) * 128 + kk * 32, 128, lane);
                acc = wmma_bf16(aq[kk], bfr, acc);
            }
            s[nt] = acc;
        }

        // online softmax (log2 domain), per row m = v + 8h of this wave's 16 rows
#pragma unroll
        for (int v = 0; v < 8; ++v) {
            float mx = -1e30f;
#pragma unroll
            for (int nt = 0; nt < 4; ++nt) { s[nt][v] *= sc; mx = fmaxf(mx, s[nt][v]); }
            mx = hmax16(mx);
            float mnew = fmaxf(m8[v], mx);
            float alpha = exp2f(m8[v] - mnew);
            m8[v] = mnew;
            float rs = 0.f;
#pragma unroll
            for (int nt = 0; nt < 4; ++nt) {
                float p = exp2f(s[nt][v] - mnew);
                s[nt][v] = p;
                rs += p;
            }
            rs = hsum16(rs);
            l8[v] = l8[v] * alpha + rs;
#pragma unroll
            for (int t = 0; t < 8; ++t) o[t][v] *= alpha;
        }

        // stage P (bf16) into this wave's private LDS rows, then relayout as A fragments
#pragma unroll
        for (int nt = 0; nt < 4; ++nt)
#pragma unroll
            for (int v = 0; v < 8; ++v)
                pl[(wave * 16 + v + 8 * h) * 64 + nt * 16 + n] = f2bf(s[nt][v]);
        asm volatile("s_wait_dscnt 0" ::: "memory");   // own LDS writes visible (in-order per wave)

        v16bf ap[2];
#pragma unroll
        for (int kk2 = 0; kk2 < 2; ++kk2)
            ap[kk2] = load_a(pl + (wave * 16) * 64 + kk2 * 32, 64, lane);

        // O += P * V   (V tile is K-major: element (j, c) at vt[j*128 + c])
#pragma unroll
        for (int t = 0; t < 8; ++t) {
#pragma unroll
            for (int kk2 = 0; kk2 < 2; ++kk2) {
                v16bf bfr = load_b_kn(vt + kk2 * 32 * 128 + t * 16, 128, lane);
                o[t] = wmma_bf16(ap[kk2], bfr, o[t]);
            }
        }
        __syncthreads();
    }

    // finalize: O /= l, store as (b, n, c) bf16
#pragma unroll
    for (int v = 0; v < 8; ++v) {
        float inv = 1.f / l8[v];
        int row = iw + v + 8 * h;
#pragma unroll
        for (int t = 0; t < 8; ++t)
            oT[base + (size_t)row * CH + t * 16 + n] = f2bf(o[t][v] * inv);
    }
}

// ---------------- kernel 6: output projection + residual ----------------
// out[b][co][i] = x[b][co][i] + gamma * ( sum_k Wo[co][k]*oT[b][i][k] + bo[co] )
__global__ void oproj_kernel(const bf16_t* __restrict__ oT, const bf16_t* __restrict__ Wob,
                             const float* __restrict__ bo, const float* __restrict__ x,
                             const float* __restrict__ gamma, float* __restrict__ out) {
    int lane = threadIdx.x & 31;
    int wave = threadIdx.x >> 5;
    int i0  = blockIdx.x * 64;
    int co0 = blockIdx.y * 64 + wave * 16;
    int b   = blockIdx.z;
    size_t basec = (size_t)b * NPIX * CH;   // oT base
    size_t basex = (size_t)b * CH * NPIX;   // x / out base
    float g = gamma[0];

    v16bf a[4];
#pragma unroll
    for (int kk = 0; kk < 4; ++kk)
        a[kk] = load_a(Wob + (size_t)co0 * CH + kk * 32, CH, lane);

    int n = lane & 15, h = lane >> 4;
#pragma unroll
    for (int nt = 0; nt < 4; ++nt) {
        v8f acc = {};
#pragma unroll
        for (int kk = 0; kk < 4; ++kk) {
            v16bf bfr = load_b_nk(oT + basec + (size_t)(i0 + nt * 16) * CH + kk * 32, CH, lane);
            acc = wmma_bf16(a[kk], bfr, acc);
        }
        int i = i0 + nt * 16 + n;
#pragma unroll
        for (int v = 0; v < 8; ++v) {
            int co = co0 + v + 8 * h;
            size_t idx = basex + (size_t)co * NPIX + i;
            out[idx] = x[idx] + g * (acc[v] + bo[co]);
        }
    }
}

// ---------------- launch ----------------
extern "C" void kernel_launch(void* const* d_in, const int* in_sizes, int n_in,
                              void* d_out, int out_size, void* d_ws, size_t ws_size,
                              hipStream_t stream) {
    const float* x  = (const float*)d_in[0];
    const float* gw = (const float*)d_in[1];
    const float* gb = (const float*)d_in[2];
    const float* wq = (const float*)d_in[3];
    const float* bq = (const float*)d_in[4];
    const float* wk = (const float*)d_in[5];
    const float* bk = (const float*)d_in[6];
    const float* wv = (const float*)d_in[7];
    const float* bv = (const float*)d_in[8];
    const float* wo = (const float*)d_in[9];
    const float* bo = (const float*)d_in[10];
    const float* gamma = (const float*)d_in[11];
    float* out = (float*)d_out;

    char* ws = (char*)d_ws;
    size_t off = 0;
    auto take = [&](size_t bytes) -> char* {
        char* p = ws + off;
        off = (off + bytes + 255) & ~(size_t)255;
        return p;
    };
    const size_t nW  = (size_t)CH * CH;            // 16384
    const size_t nAct = (size_t)BATCH * NPIX * CH; // 4 Mi elements

    float*  stats = (float*)take(BATCH * NGRP * 2 * sizeof(float));
    bf16_t* wqb = (bf16_t*)take(nW * sizeof(bf16_t));
    bf16_t* wkb = (bf16_t*)take(nW * sizeof(bf16_t));
    bf16_t* wvb = (bf16_t*)take(nW * sizeof(bf16_t));
    bf16_t* wob = (bf16_t*)take(nW * sizeof(bf16_t));
    bf16_t* xnT = (bf16_t*)take(nAct * sizeof(bf16_t));
    bf16_t* qT  = (bf16_t*)take(nAct * sizeof(bf16_t));
    bf16_t* kTp = (bf16_t*)take(nAct * sizeof(bf16_t));
    bf16_t* vTp = (bf16_t*)take(nAct * sizeof(bf16_t));
    bf16_t* oT  = (bf16_t*)take(nAct * sizeof(bf16_t));

    gn_stats_kernel<<<BATCH * NGRP, 256, 0, stream>>>(x, stats);
    gn_apply_kernel<<<dim3(NPIX / 32, BATCH), 256, 0, stream>>>(x, gw, gb, stats, xnT);

    w2bf_kernel<<<(nW + 255) / 256, 256, 0, stream>>>(wq, wqb, (int)nW);
    w2bf_kernel<<<(nW + 255) / 256, 256, 0, stream>>>(wk, wkb, (int)nW);
    w2bf_kernel<<<(nW + 255) / 256, 256, 0, stream>>>(wv, wvb, (int)nW);
    w2bf_kernel<<<(nW + 255) / 256, 256, 0, stream>>>(wo, wob, (int)nW);

    dim3 pg(BATCH * NPIX / 64, CH / 64);
    proj_kernel<<<pg, 128, 0, stream>>>(xnT, wqb, bq, qT);
    proj_kernel<<<pg, 128, 0, stream>>>(xnT, wkb, bk, kTp);
    proj_kernel<<<pg, 128, 0, stream>>>(xnT, wvb, bv, vTp);

    attn_kernel<<<dim3(NPIX / 64, BATCH), 128, 0, stream>>>(qT, kTp, vTp, oT);

    oproj_kernel<<<dim3(NPIX / 64, CH / 64, BATCH), 128, 0, stream>>>(oT, wob, bo, x, gamma, out);
}
